// GroupCommunication_30889404792946
// MI455X (gfx1250) — matrix operand order repaired and checked by reading, not compiled, not run on hardware
//
#include <hip/hip_runtime.h>
#include <hip/hip_bf16.h>

typedef __bf16 bf16;
typedef __attribute__((ext_vector_type(16))) __bf16 v16bf;
typedef __attribute__((ext_vector_type(8)))  float  v8f;

#define NB        8        // blocks
#define NH        2        // heads
#define HD        32       // head dim
#define DIM       1024
#define BLK_DIN   128
#define EMB       64       // per-block qkv width (NH*HD)
#define TILE      16       // tokens per wave tile
#define NTILES    2048     // 32768 tokens / 16
#define WPG       4        // waves per workgroup

#define NFRAG_QKV (3*8*4*4)   // p,n,j,kk  (K=128 -> 4 kk, N=64 -> 4 j)
#define NFRAG_WF  (8*8*2)     // n,j,kk    (K=64  -> 2 kk, N=128 -> 8 j)
#define FRAG_ELEMS 512        // 32 lanes * 16 bf16 per B fragment

// ---------------------------------------------------------------------------
// Pack f32 weights into bf16 WMMA B-fragments, one 32B chunk per lane.
// B (32x16) layout assumption (from CDNA5 ISA, 16-bit B 64x16 halved):
//   lane L holds column N = L%16 ; element e -> K = e + 8*(L/16) (+8 if e>=8)
// ---------------------------------------------------------------------------
__global__ void pack_weights_kernel(const float* __restrict__ wq,
                                    const float* __restrict__ wk,
                                    const float* __restrict__ wv,
                                    const float* __restrict__ wf,
                                    bf16* __restrict__ ws) {
    int idx = blockIdx.x * blockDim.x + threadIdx.x;
    const int total = (NFRAG_QKV + NFRAG_WF) * FRAG_ELEMS;
    if (idx >= total) return;
    int frag   = idx >> 9;        // /512
    int within = idx & 511;
    int lane   = within >> 4;
    int e      = within & 15;
    int g      = lane >> 4;       // half-wave
    int c      = lane & 15;       // column within fragment
    float val;
    if (frag < NFRAG_QKV) {
        int p  = frag / 128, rem = frag % 128;
        int n  = rem / 16,   r2  = rem % 16;
        int j  = r2 / 4,     kk  = r2 % 4;
        int K  = kk * 32 + e + (e >= 8 ? 8 : 0) + 8 * g;     // 0..127 (block din)
        int N  = j * 16 + c;               // 0..63
        const float* w = (p == 0) ? wq : (p == 1) ? wk : wv;
        val = w[(n * BLK_DIN + K) * EMB + N];
    } else {
        int fw = frag - NFRAG_QKV;
        int n  = fw / 16, rem = fw % 16;
        int j  = rem / 2, kk  = rem % 2;
        int K  = kk * 32 + e + (e >= 8 ? 8 : 0) + 8 * g;     // 0..63
        int N  = j * 16 + c;               // 0..127
        val = wf[(n * EMB + K) * BLK_DIN + N];
    }
    ws[idx] = (bf16)val;
}

// ---------------------------------------------------------------------------
// Main kernel: one wave per 16-token tile, 4 waves per workgroup.
// A (16x32 bf16) layout (CDNA5 ISA): lane L row M=L%16;
//   elements e<8  -> K = e + 8*(L/16)
//   elements e>=8 -> K = e + 8 + 8*(L/16)
// C/D (16x16 f32): lane L col N=L%16; VGPR r -> row M = r + 8*(L/16)
// ---------------------------------------------------------------------------
__launch_bounds__(32 * WPG)
__global__ void group_comm_kernel(const float* __restrict__ x,
                                  const bf16*  __restrict__ wpack,
                                  float* __restrict__ out) {
    // per-wave LDS slices: QKV (bf16, [p][token][h*8+blk][d]) + attn output
    __shared__ bf16 lds_qkv[WPG][3][TILE][16][HD];   // 4*48KB = 192KB
    __shared__ bf16 lds_ao [WPG][TILE][NB * EMB];    // 4*16KB =  64KB

    const int lane = threadIdx.x & 31;
    const int wave = threadIdx.x >> 5;
    const int tile = blockIdx.x * WPG + wave;
    const int c16  = lane & 15;
    const int g    = lane >> 4;

    // ---------------- Phase 1: Q,K,V projections (bf16 WMMA) ----------------
    {
        const float* xrow = x + (size_t)(tile * TILE + c16) * DIM; // A row per lane
        for (int n = 0; n < NB; ++n) {
            v16bf A[4];
            #pragma unroll
            for (int kk = 0; kk < 4; ++kk) {
                const float* p0 = xrow + n * BLK_DIN + kk * 32 + 8 * g;
                #pragma unroll
                for (int i = 0; i < 8; ++i) {
                    A[kk][i]     = (bf16)p0[i];        // K = e + 8g
                    A[kk][8 + i] = (bf16)p0[16 + i];   // K = e + 8 + 8g
                }
            }
            for (int p = 0; p < 3; ++p) {
                #pragma unroll
                for (int j = 0; j < 4; ++j) {
                    v8f acc = {};
                    #pragma unroll
                    for (int kk = 0; kk < 4; ++kk) {
                        int frag = ((p * 8 + n) * 4 + j) * 4 + kk;
                        v16bf B = *(const v16bf*)(wpack + (size_t)frag * FRAG_ELEMS + lane * 16);
                        acc = __builtin_amdgcn_wmma_f32_16x16x32_bf16(
                                  false, A[kk], false, B, (short)0, acc, false, false);
                    }
                    // scatter C to LDS: feature f = j*16 + c16 -> head, dim
                    int f = j * 16 + c16;
                    int h = f >> 5, d = f & 31;
                    #pragma unroll
                    for (int r = 0; r < 8; ++r) {
                        int t = r + 8 * g;
                        lds_qkv[wave][p][t][h * 8 + n][d] = (bf16)acc[r];
                    }
                }
            }
        }
    }

    // ---------------- Phase 2: attention, one (token, head) per lane --------
    {
        const int t = c16;      // token in tile
        const int h = g;        // head
        const bf16* qb = &lds_qkv[wave][0][t][h * 8][0];
        const bf16* kb = &lds_qkv[wave][1][t][h * 8][0];
        const bf16* vb = &lds_qkv[wave][2][t][h * 8][0];
        const float scale = 0.1767766952966369f;   // 32^-0.5

        for (int nq = 0; nq < NB; ++nq) {
            float qv[HD];
            #pragma unroll
            for (int d = 0; d < HD; ++d) qv[d] = (float)qb[nq * HD + d] * scale;
            float s[NB];
            #pragma unroll
            for (int nk = 0; nk < NB; ++nk) {
                float a = 0.f;
                #pragma unroll
                for (int d = 0; d < HD; ++d) a += qv[d] * (float)kb[nk * HD + d];
                s[nk] = a;
            }
            float m = s[0];
            #pragma unroll
            for (int nk = 1; nk < NB; ++nk) m = fmaxf(m, s[nk]);
            float sum = 0.f;
            #pragma unroll
            for (int nk = 0; nk < NB; ++nk) { s[nk] = __expf(s[nk] - m); sum += s[nk]; }
            float rinv = 1.0f / sum;
            float o[HD];
            #pragma unroll
            for (int d = 0; d < HD; ++d) o[d] = 0.f;
            #pragma unroll
            for (int nk = 0; nk < NB; ++nk) {
                #pragma unroll
                for (int d = 0; d < HD; ++d) o[d] += s[nk] * (float)vb[nk * HD + d];
            }
            #pragma unroll
            for (int d = 0; d < HD; ++d)
                lds_ao[wave][t][nq * EMB + h * HD + d] = (bf16)(o[d] * rinv);
        }
    }

    // ---------------- Phase 3: output projection wf (bf16 WMMA) -------------
    {
        float* orow = out + (size_t)(tile * TILE) * DIM;
        for (int n = 0; n < NB; ++n) {
            v16bf A[2];
            #pragma unroll
            for (int kk = 0; kk < 2; ++kk) {
                const bf16* p0 = &lds_ao[wave][c16][n * EMB + kk * 32 + 8 * g];
                #pragma unroll
                for (int i = 0; i < 8; ++i) {
                    A[kk][i]     = p0[i];
                    A[kk][8 + i] = p0[16 + i];
                }
            }
            #pragma unroll
            for (int j = 0; j < 8; ++j) {
                v8f acc = {};
                #pragma unroll
                for (int kk = 0; kk < 2; ++kk) {
                    int frag = NFRAG_QKV + (n * 8 + j) * 2 + kk;
                    v16bf B = *(const v16bf*)(wpack + (size_t)frag * FRAG_ELEMS + lane * 16);
                    acc = __builtin_amdgcn_wmma_f32_16x16x32_bf16(
                              false, A[kk], false, B, (short)0, acc, false, false);
                }
                #pragma unroll
                for (int r = 0; r < 8; ++r) {
                    int t = r + 8 * g;
                    orow[(size_t)t * DIM + n * BLK_DIN + j * 16 + c16] = acc[r];
                }
            }
        }
    }
}

// ---------------------------------------------------------------------------
extern "C" void kernel_launch(void* const* d_in, const int* in_sizes, int n_in,
                              void* d_out, int out_size, void* d_ws, size_t ws_size,
                              hipStream_t stream) {
    const float* x  = (const float*)d_in[0];
    const float* wq = (const float*)d_in[1];
    const float* wk = (const float*)d_in[2];
    const float* wv = (const float*)d_in[3];
    const float* wf = (const float*)d_in[4];
    bf16* ws = (bf16*)d_ws;   // needs (384+128)*512*2 = 512KB of workspace

    const int total = (NFRAG_QKV + NFRAG_WF) * FRAG_ELEMS;
    pack_weights_kernel<<<(total + 255) / 256, 256, 0, stream>>>(wq, wk, wv, wf, ws);

    group_comm_kernel<<<NTILES / WPG, 32 * WPG, 0, stream>>>(x, ws, (float*)d_out);
}
